// MultiHeadAttention_16363825397851
// MI455X (gfx1250) — matrix operand order repaired and checked by reading, not compiled, not run on hardware
//
#include <hip/hip_runtime.h>

// ---------------------------------------------------------------------------
// MHA pipeline for MI455X (gfx1250, wave32, WMMA, async-to-LDS)
//   S=2048, D=1024, H=16
//   All GEMMs: bf16 operands, f32 accumulation, v_wmma_f32_16x16x32_bf16.
//   Every GEMM consumes B in [N][K] layout so LDS staging is pure b128 copies.
// ---------------------------------------------------------------------------

typedef __attribute__((ext_vector_type(16))) __bf16 v16bf;
typedef __attribute__((ext_vector_type(8)))  float  v8f;

#define S_DIM 2048
#define D_DIM 1024
#define H_DIM 16

#if defined(__AMDGCN__) && __has_builtin(__builtin_amdgcn_global_load_async_to_lds_b128)
#define USE_ASYNC_LDS 1
#else
#define USE_ASYNC_LDS 0
#endif

__device__ __forceinline__ unsigned short f32_to_bf16_rte(float f) {
    unsigned int u = __builtin_bit_cast(unsigned int, f);
    unsigned int r = u + 0x7FFFu + ((u >> 16) & 1u);  // round-to-nearest-even
    return (unsigned short)(r >> 16);
}

__device__ __forceinline__ void wait_async_lds() {
#if USE_ASYNC_LDS
#if __has_builtin(__builtin_amdgcn_s_wait_asynccnt)
    __builtin_amdgcn_s_wait_asynccnt(0);
#else
    asm volatile("s_wait_asynccnt 0" ::: "memory");
#endif
#endif
}

#if USE_ASYNC_LDS
// Builtin signature (from compiler diagnostic): (v4i addrspace(1)*, v4i addrspace(3)*, imm, imm)
typedef int v4i_gcc __attribute__((vector_size(16)));
typedef __attribute__((address_space(1))) v4i_gcc gv4i;
typedef __attribute__((address_space(3))) v4i_gcc lv4i;
__device__ __forceinline__ void async_b128(const void* g, void* l) {
    __builtin_amdgcn_global_load_async_to_lds_b128((gv4i*)g, (lv4i*)l, 0, 0);
}
#endif

// ---------------------------------------------------------------------------
// f32 -> bf16 streaming convert (row-major preserved)
// ---------------------------------------------------------------------------
__global__ __launch_bounds__(256) void cvt_f32_bf16(const float* __restrict__ src,
                                                    unsigned short* __restrict__ dst,
                                                    long long n) {
    long long i = (long long)blockIdx.x * blockDim.x + threadIdx.x;
    if (i < n) dst[i] = f32_to_bf16_rte(src[i]);
}

// ---------------------------------------------------------------------------
// f32 -> bf16 convert + transpose: dst[b][c][r] = bf16(src[b][r][c])
// 32x32 tiles, block (32,8)
// ---------------------------------------------------------------------------
__global__ __launch_bounds__(256) void cvt_f32_bf16_T(const float* __restrict__ src,
                                                      unsigned short* __restrict__ dst,
                                                      int R, int C) {
    __shared__ float tile[32][33];
    const long long bo = (long long)blockIdx.z * R * C;
    const int r0 = blockIdx.y * 32, c0 = blockIdx.x * 32;
    const int tx = threadIdx.x, ty = threadIdx.y;
#pragma unroll
    for (int j = 0; j < 4; ++j)
        tile[ty + j * 8][tx] = src[bo + (long long)(r0 + ty + j * 8) * C + (c0 + tx)];
    __syncthreads();
#pragma unroll
    for (int j = 0; j < 4; ++j)
        dst[bo + (long long)(c0 + ty + j * 8) * R + (r0 + tx)] =
            f32_to_bf16_rte(tile[tx][ty + j * 8]);
}

// ---------------------------------------------------------------------------
// Tiled bf16 WMMA GEMM:  C[b] = A[b] (MxK, row-major) * B[b]^T (B is NxK)
// BM=256, BN=128, BK=32; 256 threads = 8 waves (4 along M x 2 along N),
// each wave owns a 64x64 C tile: 16 accumulators, 16 WMMAs per K-step.
// Double-buffered LDS; async global->LDS when available.
// STORE: 0 = bf16 row-major C[row*N+col]
//        1 = bf16 transposed C[col*ldcT + row]   (batch offset bz*sC)
//        2 = f32 row-major (+ optional bias)
// ---------------------------------------------------------------------------
template <int STORE>
__global__ __launch_bounds__(256) void gemm_bf16_wmma(
    const unsigned short* __restrict__ A,
    const unsigned short* __restrict__ B,
    void* __restrict__ Cv,
    const float* __restrict__ bias,
    int M, int N, int Kd, int ldcT,
    long long sA, long long sB, long long sC)
{
    constexpr int BM = 256, BN = 128, BK = 32, PAD = 8, LDP = BK + PAD;
    __shared__ __align__(16) unsigned short As[2][BM][LDP];
    __shared__ __align__(16) unsigned short Bs[2][BN][LDP];

    const int bz = blockIdx.z;
    const unsigned short* Ab = A + (long long)bz * sA;
    const unsigned short* Bb = B + (long long)bz * sB;

    const int m0 = blockIdx.x * BM;
    const int n0 = blockIdx.y * BN;
    const int t    = threadIdx.x;
    const int wave = t >> 5;
    const int lane = t & 31;
    const int wm   = (wave & 3) * 64;   // 4 waves along M
    const int wn   = (wave >> 2) * 64;  // 2 waves along N
    const int lrow = lane & 15;
    const int lg   = lane >> 4;

    // per-thread staging coordinates (b128 granules)
    // A tile: 1024 granules -> 4/thread ; B tile: 512 granules -> 2/thread
    int arow[4], acol[4], brow[2], bcol[2];
#pragma unroll
    for (int j = 0; j < 4; ++j) { int c = t + j * 256; arow[j] = c >> 2; acol[j] = (c & 3) * 8; }
#pragma unroll
    for (int j = 0; j < 2; ++j) { int c = t + j * 256; brow[j] = c >> 2; bcol[j] = (c & 3) * 8; }

#if !USE_ASYNC_LDS
    uint4 ra[4], rb[2];
#endif

    // ---- prologue: stage tile k0=0 into buffer 0
#if USE_ASYNC_LDS
#pragma unroll
    for (int j = 0; j < 4; ++j)
        async_b128(Ab + (long long)(m0 + arow[j]) * Kd + acol[j], &As[0][arow[j]][acol[j]]);
#pragma unroll
    for (int j = 0; j < 2; ++j)
        async_b128(Bb + (long long)(n0 + brow[j]) * Kd + bcol[j], &Bs[0][brow[j]][bcol[j]]);
    wait_async_lds();
#else
#pragma unroll
    for (int j = 0; j < 4; ++j)
        ra[j] = *reinterpret_cast<const uint4*>(Ab + (long long)(m0 + arow[j]) * Kd + acol[j]);
#pragma unroll
    for (int j = 0; j < 2; ++j)
        rb[j] = *reinterpret_cast<const uint4*>(Bb + (long long)(n0 + brow[j]) * Kd + bcol[j]);
#pragma unroll
    for (int j = 0; j < 4; ++j) *reinterpret_cast<uint4*>(&As[0][arow[j]][acol[j]]) = ra[j];
#pragma unroll
    for (int j = 0; j < 2; ++j) *reinterpret_cast<uint4*>(&Bs[0][brow[j]][bcol[j]]) = rb[j];
#endif
    __syncthreads();

    v8f acc[4][4] = {};

    for (int k0 = 0; k0 < Kd; k0 += BK) {
        const int buf = (k0 / BK) & 1;
        const bool has_next = (k0 + BK) < Kd;
        const int kn = k0 + BK;

        // ---- kick off staging of the next tile into the other buffer
        if (has_next) {
#if USE_ASYNC_LDS
#pragma unroll
            for (int j = 0; j < 4; ++j)
                async_b128(Ab + (long long)(m0 + arow[j]) * Kd + kn + acol[j],
                           &As[buf ^ 1][arow[j]][acol[j]]);
#pragma unroll
            for (int j = 0; j < 2; ++j)
                async_b128(Bb + (long long)(n0 + brow[j]) * Kd + kn + bcol[j],
                           &Bs[buf ^ 1][brow[j]][bcol[j]]);
#else
#pragma unroll
            for (int j = 0; j < 4; ++j)
                ra[j] = *reinterpret_cast<const uint4*>(
                    Ab + (long long)(m0 + arow[j]) * Kd + kn + acol[j]);
#pragma unroll
            for (int j = 0; j < 2; ++j)
                rb[j] = *reinterpret_cast<const uint4*>(
                    Bb + (long long)(n0 + brow[j]) * Kd + kn + bcol[j]);
#endif
        }

        // ---- fragments (CDNA5 16-bit A 16x32 / B 32x16 lane layouts)
        union Frag { uint4 q[2]; v16bf v; };
        Frag a[4], b[4];
#pragma unroll
        for (int mi = 0; mi < 4; ++mi) {
            const unsigned short* ap = &As[buf][wm + mi * 16 + lrow][0];
            a[mi].q[0] = *reinterpret_cast<const uint4*>(ap + lg * 8);
            a[mi].q[1] = *reinterpret_cast<const uint4*>(ap + 16 + lg * 8);
        }
#pragma unroll
        for (int ni = 0; ni < 4; ++ni) {
            const unsigned short* bp = &Bs[buf][wn + ni * 16 + lrow][0];
            b[ni].q[0] = *reinterpret_cast<const uint4*>(bp + lg * 8);
            b[ni].q[1] = *reinterpret_cast<const uint4*>(bp + 16 + lg * 8);
        }
#pragma unroll
        for (int mi = 0; mi < 4; ++mi)
#pragma unroll
            for (int ni = 0; ni < 4; ++ni)
                acc[mi][ni] = __builtin_amdgcn_wmma_f32_16x16x32_bf16(
                    false, a[mi].v, false, b[ni].v,
                    (short)0, acc[mi][ni], false, false);

        // ---- finish staging, flip buffers
#if USE_ASYNC_LDS
        wait_async_lds();
#else
        if (has_next) {
#pragma unroll
            for (int j = 0; j < 4; ++j)
                *reinterpret_cast<uint4*>(&As[buf ^ 1][arow[j]][acol[j]]) = ra[j];
#pragma unroll
            for (int j = 0; j < 2; ++j)
                *reinterpret_cast<uint4*>(&Bs[buf ^ 1][brow[j]][bcol[j]]) = rb[j];
        }
#endif
        __syncthreads();
    }

    // ---- store: C/D layout col = lane%16, M = lg*8 + r
#pragma unroll
    for (int mi = 0; mi < 4; ++mi) {
#pragma unroll
        for (int ni = 0; ni < 4; ++ni) {
#pragma unroll
            for (int r = 0; r < 8; ++r) {
                int row = m0 + wm + mi * 16 + lg * 8 + r;
                int col = n0 + wn + ni * 16 + lrow;
                float val = acc[mi][ni][r];
                if (STORE == 0) {
                    unsigned short* C = (unsigned short*)Cv + (long long)bz * sC;
                    C[(long long)row * N + col] = f32_to_bf16_rte(val);
                } else if (STORE == 1) {
                    unsigned short* C = (unsigned short*)Cv;
                    C[(long long)bz * sC + (long long)col * ldcT + row] = f32_to_bf16_rte(val);
                } else {
                    float* C = (float*)Cv + (long long)bz * sC;
                    if (bias) val += bias[(long long)row * N + col];
                    C[(long long)row * N + col] = val;
                }
            }
        }
    }
}

// ---------------------------------------------------------------------------
// Row-wise softmax: attn = softmax(scale * scores), f32 in -> bf16 out.
// ---------------------------------------------------------------------------
__global__ __launch_bounds__(256) void softmax_rows(const float* __restrict__ scores,
                                                    unsigned short* __restrict__ attn,
                                                    int len, float scale) {
    long long row = blockIdx.x;
    const float* x = scores + row * len;
    unsigned short* y = attn + row * len;
    __shared__ float red[256];
    int t = threadIdx.x;

    float m = -1e30f;
    for (int i = t; i < len; i += 256) m = fmaxf(m, x[i] * scale);
    red[t] = m; __syncthreads();
    for (int s = 128; s > 0; s >>= 1) {
        if (t < s) red[t] = fmaxf(red[t], red[t + s]);
        __syncthreads();
    }
    m = red[0]; __syncthreads();

    float sum = 0.0f;
    for (int i = t; i < len; i += 256) sum += __expf(x[i] * scale - m);
    red[t] = sum; __syncthreads();
    for (int s = 128; s > 0; s >>= 1) {
        if (t < s) red[t] += red[t + s];
        __syncthreads();
    }
    float inv = 1.0f / red[0];

    for (int i = t; i < len; i += 256)
        y[i] = f32_to_bf16_rte(__expf(x[i] * scale - m) * inv);
}

// ---------------------------------------------------------------------------
// Orchestration
// ---------------------------------------------------------------------------
extern "C" void kernel_launch(void* const* d_in, const int* in_sizes, int n_in,
                              void* d_out, int out_size, void* d_ws, size_t ws_size,
                              hipStream_t stream) {
    const int S = S_DIM, D = D_DIM, H = H_DIM;
    const float scale = 0.03125f;  // 1/sqrt(1024)

    const float* data  = (const float*)d_in[0];   // [S,D]
    const float* wq    = (const float*)d_in[1];   // [H,D,D]
    const float* wk    = (const float*)d_in[2];
    const float* wv    = (const float*)d_in[3];
    const float* w_lin = (const float*)d_in[4];   // [S, S*H]
    const float* b_lin = (const float*)d_in[5];   // [S,D]
    float* out = (float*)d_out;                   // [S,D]

    // ---- workspace layout --------------------------------------------------
    char* ws = (char*)d_ws;
    size_t off = 0;
    auto alloc = [&](size_t bytes) { char* p = ws + off; off += (bytes + 255) & ~size_t(255); return p; };

    unsigned short* data_bf = (unsigned short*)alloc((size_t)S * D * 2);       //   4 MB
    unsigned short* wqT_bf  = (unsigned short*)alloc((size_t)H * D * D * 2);   //  32 MB  [H][E][D]
    unsigned short* wkT_bf  = (unsigned short*)alloc((size_t)H * D * D * 2);   //  32 MB
    unsigned short* wvT_bf  = (unsigned short*)alloc((size_t)H * D * D * 2);   //  32 MB
    unsigned short* wl_bf   = (unsigned short*)alloc((size_t)S * S * H * 2);   // 128 MB
    unsigned short* Qb      = (unsigned short*)alloc((size_t)H * S * D * 2);   //  64 MB  [H][S][E]
    unsigned short* Kb      = (unsigned short*)alloc((size_t)H * S * D * 2);   //  64 MB  [H][T][E]
    unsigned short* VbT     = (unsigned short*)alloc((size_t)H * S * D * 2);   //  64 MB  [H][E][T]
    unsigned short* zT      = (unsigned short*)alloc((size_t)D * S * H * 2);   //  64 MB  [E][H*S]
    float*          scoresb = (float*)alloc((size_t)H * S * S * 4);            // 256 MB
    unsigned short* attnb   = (unsigned short*)alloc((size_t)H * S * S * 2);   // 128 MB
    (void)ws_size;

    // ---- 1) convert inputs to bf16 (weights transposed to [N][K]) ---------
    {
        long long n1 = (long long)S * D;
        cvt_f32_bf16<<<(unsigned)((n1 + 255) / 256), 256, 0, stream>>>(data, data_bf, n1);
        long long n2 = (long long)S * S * H;
        cvt_f32_bf16<<<(unsigned)((n2 + 255) / 256), 256, 0, stream>>>(w_lin, wl_bf, n2);
        dim3 tg(D / 32, D / 32, H), tb(32, 8);
        cvt_f32_bf16_T<<<tg, tb, 0, stream>>>(wq, wqT_bf, D, D);
        cvt_f32_bf16_T<<<tg, tb, 0, stream>>>(wk, wkT_bf, D, D);
        cvt_f32_bf16_T<<<tg, tb, 0, stream>>>(wv, wvT_bf, D, D);
    }

    // ---- 2) Q/K/V projections: data [S,D] x W^T[h] ([E][D]) ---------------
    {
        dim3 grid(S / 256, D / 128, H), block(256);
        gemm_bf16_wmma<0><<<grid, block, 0, stream>>>(
            data_bf, wqT_bf, Qb, nullptr, S, D, D, 0,
            0, (long long)D * D, (long long)S * D);
        gemm_bf16_wmma<0><<<grid, block, 0, stream>>>(
            data_bf, wkT_bf, Kb, nullptr, S, D, D, 0,
            0, (long long)D * D, (long long)S * D);
        // V stored transposed: VbT[h][e][t] = v[h][t][e]
        gemm_bf16_wmma<1><<<grid, block, 0, stream>>>(
            data_bf, wvT_bf, VbT, nullptr, S, D, D, /*ldcT=*/S,
            0, (long long)D * D, (long long)S * D);
    }

    // ---- 3) scores = Q @ K^T (f32 out; scale folded into softmax) ---------
    {
        dim3 grid(S / 256, S / 128, H), block(256);
        gemm_bf16_wmma<2><<<grid, block, 0, stream>>>(
            Qb, Kb, scoresb, nullptr, S, S, D, 0,
            (long long)S * D, (long long)S * D, (long long)S * S);
    }

    // ---- 4) softmax rows ---------------------------------------------------
    softmax_rows<<<H * S, 256, 0, stream>>>(scoresb, attnb, S, scale);

    // ---- 5) z = attn @ V, stored transposed: zT[e][h*S+s] ------------------
    {
        dim3 grid(S / 256, D / 128, H), block(256);
        gemm_bf16_wmma<1><<<grid, block, 0, stream>>>(
            attnb, VbT, zT, nullptr, S, D, S, /*ldcT=*/H * S,
            (long long)S * S, (long long)S * D, (long long)S);
    }

    // ---- 6) out = w_lin @ concat(z) + b_lin (f32) --------------------------
    {
        dim3 grid(S / 256, D / 128, 1), block(256);
        gemm_bf16_wmma<2><<<grid, block, 0, stream>>>(
            wl_bf, zT, out, b_lin, S, D, S * H, 0,
            0, 0, 0);
    }
    (void)in_sizes; (void)n_in; (void)out_size;
}
